// Node_PointCloudEncoder_80788334838051
// MI455X (gfx1250) — matrix operand order repaired and checked by reading, not compiled, not run on hardware
//
#include <hip/hip_runtime.h>
#include <hip/hip_bf16.h>

typedef __attribute__((ext_vector_type(16))) _Float16 v16h;
typedef __attribute__((ext_vector_type(8)))  _Float16 v8h;
typedef __attribute__((ext_vector_type(2)))  _Float16 h2;
typedef __attribute__((ext_vector_type(8)))  float    v8f;

#define HD     128     // hidden dim
#define TILE_P 256     // points per block
#define ROWP   136     // padded LDS row length in halves (272B, 16B multiple, bank-spread)

// ---------------- zero the output (scatter-max target must start at 0) ------
__global__ void zero_out_kernel(float4* __restrict__ out4, int n4) {
    int i = blockIdx.x * blockDim.x + threadIdx.x;
    if (i < n4) out4[i] = make_float4(0.f, 0.f, 0.f, 0.f);
}

// -------- fold BN into layer-1 (emit f16), transpose W2 -> f16 [n][k] -------
__global__ void prep_kernel(const float* __restrict__ W1, const float* __restrict__ b1,
                            const float* __restrict__ gamma, const float* __restrict__ beta,
                            const float* __restrict__ rmean, const float* __restrict__ rvar,
                            const float* __restrict__ W2,
                            _Float16* __restrict__ W1h, _Float16* __restrict__ b1h,
                            _Float16* __restrict__ W2t) {
    int j = threadIdx.x;                      // 0..127
    float sc = gamma[j] * rsqrtf(rvar[j] + 1e-5f);
    W1h[0 * HD + j] = (_Float16)(W1[0 * HD + j] * sc);
    W1h[1 * HD + j] = (_Float16)(W1[1 * HD + j] * sc);
    W1h[2 * HD + j] = (_Float16)(W1[2 * HD + j] * sc);
    b1h[j] = (_Float16)((b1[j] - rmean[j]) * sc + beta[j]);
    // W2 is [k][n] row-major; store column j contiguously over k for B-fragments
    for (int k = 0; k < HD; ++k)
        W2t[j * HD + k] = (_Float16)W2[k * HD + j];
}

// ---------------- fused MLP + WMMA GEMM + branchless scatter-max ------------
__global__ __launch_bounds__(256) void encoder_kernel(
    const float* __restrict__ x, const int* __restrict__ bidx,
    const _Float16* __restrict__ W1h, const _Float16* __restrict__ b1h,
    const _Float16* __restrict__ W2t, const float* __restrict__ b2,
    float* __restrict__ out, int P)
{
    __shared__ alignas(16) _Float16 hbuf[TILE_P * ROWP];   // 68 KB activation tile (f16)
    __shared__ alignas(16) int nodeb[TILE_P];

    const int t    = threadIdx.x;
    const int base = blockIdx.x * TILE_P;
    const int p    = base + t;

    // ---- phase 1: layer1 in packed f16 (v_pk_fma_f16) + ReLU -> f16 LDS ----
    float x0 = 0.f, x1 = 0.f, x2 = 0.f;
    if (p < P) {
        const float* xp = x + (size_t)p * 3;
        x0 = __builtin_nontemporal_load(xp + 0);
        x1 = __builtin_nontemporal_load(xp + 1);
        x2 = __builtin_nontemporal_load(xp + 2);
        nodeb[t] = __builtin_nontemporal_load(&bidx[p]);
    } else {
        nodeb[t] = -1;                        // invalid point: neutralized in epilogue
    }
    const _Float16 xh0 = (_Float16)x0, xh1 = (_Float16)x1, xh2 = (_Float16)x2;
    const h2 X0 = { xh0, xh0 }, X1 = { xh1, xh1 }, X2 = { xh2, xh2 };

    union V8 { v8h v; h2 h[4]; };
    #pragma unroll 4
    for (int jb = 0; jb < HD / 8; ++jb) {
        const int j0 = jb * 8;
        V8 w0, w1, w2, bb, o;                 // uniform addresses -> SMEM loads
        w0.v = *(const v8h*)(W1h + 0 * HD + j0);
        w1.v = *(const v8h*)(W1h + 1 * HD + j0);
        w2.v = *(const v8h*)(W1h + 2 * HD + j0);
        bb.v = *(const v8h*)(b1h + j0);
        #pragma unroll
        for (int q = 0; q < 4; ++q) {
            h2 acc = __builtin_elementwise_fma(X2, w2.h[q], bb.h[q]);
            acc    = __builtin_elementwise_fma(X1, w1.h[q], acc);
            acc    = __builtin_elementwise_fma(X0, w0.h[q], acc);
            h2 z = {};
            o.h[q] = __builtin_elementwise_max(acc, z);   // v_pk_max_num_f16
        }
        *(v8h*)&hbuf[t * ROWP + j0] = o.v;
    }
    __syncthreads();

    // ---- phase 2: per-wave 16-column tile of the 128x128 GEMM via WMMA ------
    const int  w    = t >> 5;                 // wave id 0..7 -> N-tile
    const int  lane = t & 31;
    const int  nloc = lane & 15;
    const int  n    = w * 16 + nloc;          // output column for B/C/D role
    const bool hi   = lane >= 16;
    const int  aoff = hi ? 8 : 0;             // A-fragment K sub-offset per ISA layout

    // B fragments: lane holds column n, K = 8h+s and 16+8h+s (s<8) per K-step
    v16h B[4];
    #pragma unroll
    for (int ks = 0; ks < 4; ++ks) {
        const _Float16* bp = W2t + n * HD + ks * 32 + (hi ? 8 : 0);
        v8h lo = *(const v8h*)bp;
        v8h hh = *(const v8h*)(bp + 16);
        B[ks] = __builtin_shufflevector(lo, hh, 0,1,2,3,4,5,6,7,8,9,10,11,12,13,14,15);
    }
    const float bias = b2[n];
    unsigned* __restrict__ outu = (unsigned*)out;   // uniform base; 32-bit offsets

    #pragma unroll 1
    for (int mt = 0; mt < TILE_P / 16; ++mt) {
        const _Float16* arow = &hbuf[(mt * 16 + nloc) * ROWP];  // lane%16 = A row
        v8f c = {};
        #pragma unroll
        for (int ks = 0; ks < 4; ++ks) {
            v8h alo = *(const v8h*)(arow + ks * 32 + aoff);
            v8h ahi = *(const v8h*)(arow + ks * 32 + 16 + aoff);
            v16h a  = __builtin_shufflevector(alo, ahi,
                         0,1,2,3,4,5,6,7,8,9,10,11,12,13,14,15);
            c = __builtin_amdgcn_wmma_f32_16x16x32_f16(
                    false, a, false, B[ks], (short)0, c, false, false);
        }
        // preload the 8 node ids for this tile (two ds_load_b128, single wait)
        const int rbase = mt * 16 + (hi ? 8 : 0);           // 32B aligned
        int4 na = *(const int4*)&nodeb[rbase];
        int4 nb = *(const int4*)&nodeb[rbase + 4];
        int nds[8] = { na.x, na.y, na.z, na.w, nb.x, nb.y, nb.z, nb.w };

        // branchless epilogue: bias + ReLU + scatter-max.
        // out starts at 0 and holds only values >= 0, so max(out[*], 0) is a
        // no-op -> invalid lanes redirect to node 0 with value 0.
        #pragma unroll
        for (int r = 0; r < 8; ++r) {
            int   node = nds[r];
            float v    = fmaxf(c[r] + bias, 0.f);
            v          = node >= 0 ? v : 0.f;
            unsigned off = (((unsigned)(node > 0 ? node : 0)) << 7) + (unsigned)n;
            atomicMax(outu + off, __float_as_uint(v));
        }
    }
}

extern "C" void kernel_launch(void* const* d_in, const int* in_sizes, int n_in,
                              void* d_out, int out_size, void* d_ws, size_t ws_size,
                              hipStream_t stream) {
    const float* x     = (const float*)d_in[0];
    const int*   bidx  = (const int*)  d_in[1];
    // d_in[2] = num_nodes scalar (implied by out_size, unused on device)
    const float* W1    = (const float*)d_in[3];
    const float* b1    = (const float*)d_in[4];
    const float* gamma = (const float*)d_in[5];
    const float* beta  = (const float*)d_in[6];
    const float* rmean = (const float*)d_in[7];
    const float* rvar  = (const float*)d_in[8];
    const float* W2    = (const float*)d_in[9];
    const float* b2    = (const float*)d_in[10];
    float* out = (float*)d_out;
    const int P = in_sizes[1];                 // batch_idx length = #points

    char* ws = (char*)d_ws;
    _Float16* W2t = (_Float16*)ws;             // 128*128 f16 (transposed)      32768 B
    _Float16* W1h = (_Float16*)(ws + 32768);   // 3*128 f16 (BN-folded)           768 B
    _Float16* b1h = (_Float16*)(ws + 32768 + 768);  // 128 f16                    256 B

    int n4 = out_size / 4;                     // 20000*128 divisible by 4
    zero_out_kernel<<<(n4 + 255) / 256, 256, 0, stream>>>((float4*)d_out, n4);
    prep_kernel<<<1, 128, 0, stream>>>(W1, b1, gamma, beta, rmean, rvar, W2,
                                       W1h, b1h, W2t);
    int nblk = (P + TILE_P - 1) / TILE_P;
    encoder_kernel<<<nblk, 256, 0, stream>>>(x, bidx, W1h, b1h, W2t, b2, out, P);
}
